// Attention_21028159881295
// MI455X (gfx1250) — matrix operand order repaired and checked by reading, not compiled, not run on hardware
//
#include <hip/hip_runtime.h>
#include <hip/hip_bf16.h>

typedef __attribute__((ext_vector_type(16))) _Float16 v16h;
typedef __attribute__((ext_vector_type(8)))  _Float16 v8h;
typedef __attribute__((ext_vector_type(8)))  float    v8f;
typedef __attribute__((ext_vector_type(4)))  int      v4i;

#define WMMA_F16(a, b, c) \
  __builtin_amdgcn_wmma_f32_16x16x32_f16(false, (a), false, (b), (short)0, (c), false, false)

// ---------------------------------------------------------------------------
// Async global->LDS copy (CDNA5 GLOBAL_LOAD_ASYNC_TO_LDS_B128, ASYNCcnt),
// with a synchronous VGPR-round-trip fallback if the builtin is absent.
// Builtin signature (probe-confirmed): (v4i as1*, v4i as3*, Ii, Ii).
// ---------------------------------------------------------------------------
#if defined(__has_builtin)
#if __has_builtin(__builtin_amdgcn_global_load_async_to_lds_b128)
#define HAVE_ASYNC_LDS 1
#endif
#if __has_builtin(__builtin_amdgcn_permlane16)
#define HAVE_PERMLANE16 1
#endif
#endif
#ifndef HAVE_ASYNC_LDS
#define HAVE_ASYNC_LDS 0
#endif
#ifndef HAVE_PERMLANE16
#define HAVE_PERMLANE16 0
#endif

typedef __attribute__((address_space(1))) v4i* gbl_v4i_ptr;
typedef __attribute__((address_space(3))) v4i* lds_v4i_ptr;

__device__ __forceinline__ void async_copy16(void* lds, const void* g) {
#if HAVE_ASYNC_LDS
  __builtin_amdgcn_global_load_async_to_lds_b128(
      (gbl_v4i_ptr)(uintptr_t)g,
      (lds_v4i_ptr)(unsigned)(uintptr_t)lds,
      0, 0);
#else
  *(v8h*)lds = *(const v8h*)g;
#endif
}

__device__ __forceinline__ void async_join() {
#if HAVE_ASYNC_LDS
#if __has_builtin(__builtin_amdgcn_s_wait_asynccnt)
  __builtin_amdgcn_s_wait_asynccnt(0);
#else
  asm volatile("s_wait_asynccnt 0x0" ::: "memory");
#endif
#endif
}

// xor-by-MASK permutation within each 16-lane row, on the VALU pipe
// (v_permlane16_b32) when available; ds_bpermute fallback via __shfl_xor.
constexpr unsigned xor_sel_lo(int k) {
  unsigned v = 0;
  for (int i = 0; i < 8; ++i) v |= (unsigned)((i ^ k) & 0xF) << (4 * i);
  return v;
}
constexpr unsigned xor_sel_hi(int k) {
  unsigned v = 0;
  for (int i = 8; i < 16; ++i) v |= (unsigned)((i ^ k) & 0xF) << (4 * (i - 8));
  return v;
}

template <int MASK>
__device__ __forceinline__ float xor16(float x) {
#if HAVE_PERMLANE16
  constexpr unsigned lo = xor_sel_lo(MASK);
  constexpr unsigned hi = xor_sel_hi(MASK);
  int i = __float_as_int(x);
  return __int_as_float(__builtin_amdgcn_permlane16(i, i, lo, hi, false, false));
#else
  return __shfl_xor(x, MASK, 32);
#endif
}

__device__ __forceinline__ v16h cat8(v8h lo, v8h hi) {
  v16h r;
#pragma unroll
  for (int i = 0; i < 8; ++i) { r[i] = lo[i]; r[8 + i] = hi[i]; }
  return r;
}

__device__ __forceinline__ v16h load_frag16(const _Float16* p0, const _Float16* p1) {
  return cat8(*(const v8h*)p0, *(const v8h*)p1);
}

// ---------------------------------------------------------------------------
// fp32 -> f16 conversion
// ---------------------------------------------------------------------------
__global__ void cvt_f32_to_f16(const float* __restrict__ src, _Float16* __restrict__ dst, int n) {
  for (int i = blockIdx.x * blockDim.x + threadIdx.x; i < n; i += gridDim.x * blockDim.x)
    dst[i] = (_Float16)src[i];
}

// fp32 src[K][N] -> f16 dst[N][K]  (tiled 32x32 transpose, 256 threads)
__global__ __launch_bounds__(256)
void cvt_transpose_f16(const float* __restrict__ src, _Float16* __restrict__ dst,
                       int K, int N) {
  __shared__ float tile[32][33];
  const int k0 = blockIdx.y * 32, n0 = blockIdx.x * 32;
  const int tx = threadIdx.x & 31, ty = threadIdx.x >> 5;   // ty: 0..7
#pragma unroll
  for (int i = ty; i < 32; i += 8)
    tile[i][tx] = src[(size_t)(k0 + i) * N + (n0 + tx)];
  __syncthreads();
#pragma unroll
  for (int i = ty; i < 32; i += 8)
    dst[(size_t)(n0 + i) * K + (k0 + tx)] = (_Float16)tile[tx][i];
}

// ---------------------------------------------------------------------------
// Tiled f16 WMMA GEMM: C[M,N] = A[M,K] @ Bt^T + bias, Bt stored [N][K].
// Block tile 128(M) x 128(N); 8 waves in 4(M) x 2(N); wave tile 32x64
// (2x4 WMMA accumulators). K staged in 32-chunks, double-buffered LDS with
// async global->LDS copies. K, N compile-time; M%128==0, K%32==0 assumed.
// ---------------------------------------------------------------------------
template <int K, int N, bool OUTF>
__global__ __launch_bounds__(256)
void gemm_wmma(const _Float16* __restrict__ A, const _Float16* __restrict__ Bt,
               const float* __restrict__ bias, void* __restrict__ outPtr) {
  __shared__ _Float16 As[2][128][32];   // 16 KB
  __shared__ _Float16 Bs[2][128][32];   // 16 KB  ([n][k])

  const int tid  = threadIdx.x;
  const int lane = tid & 31;
  const int w    = tid >> 5;
  const int wm   = w >> 1;             // 0..3
  const int wn   = w & 1;              // 0..1
  const int m0   = blockIdx.y * 128;
  const int n0   = blockIdx.x * 128;

  const int mloc = lane & 15;
  const int koff = (lane >> 4) << 3;   // unused; A-frag also uses kb16 (0 / 16)
  const int kb16 = (lane >> 4) << 4;   // B-frag K half-offset: 0 / 16
  const int mgrp = (lane >> 4) << 3;   // C-frag M group: 0 / 8

  v8f zero = {};
  v8f acc[2][4];
#pragma unroll
  for (int mi = 0; mi < 2; ++mi)
#pragma unroll
    for (int ni = 0; ni < 4; ++ni) acc[mi][ni] = zero;

  auto stage = [&](int k0, int buf) {
#pragma unroll
    for (int i = 0; i < 2; ++i) {                 // A: 128x32 = 512 chunks
      int c = tid + i * 256;
      int r = c >> 2, kc = (c & 3) << 3;
      async_copy16(&As[buf][r][kc], &A[(size_t)(m0 + r) * K + (k0 + kc)]);
    }
#pragma unroll
    for (int i = 0; i < 2; ++i) {                 // Bt: 128x32 = 512 chunks
      int c = tid + i * 256;
      int r = c >> 2, kc = (c & 3) << 3;
      async_copy16(&Bs[buf][r][kc], &Bt[(size_t)(n0 + r) * K + (k0 + kc)]);
    }
  };

  stage(0, 0);
  int buf = 0;
  for (int k0 = 0; k0 < K; k0 += 32, buf ^= 1) {
    async_join();
    __syncthreads();                    // all copies into buf visible; all waves
                                        // finished computing on buf^1
    if (k0 + 32 < K) stage(k0 + 32, buf ^ 1);

    v16h af[2], bf[4];
#pragma unroll
    for (int mi = 0; mi < 2; ++mi) {
      const _Float16* r = &As[buf][wm * 32 + mi * 16 + mloc][0];
      af[mi] = load_frag16(r + kb16, r + kb16 + 8);
    }
#pragma unroll
    for (int ni = 0; ni < 4; ++ni) {
      const _Float16* r = &Bs[buf][wn * 64 + ni * 16 + mloc][kb16];
      bf[ni] = load_frag16(r, r + 8);
    }
#pragma unroll
    for (int mi = 0; mi < 2; ++mi)
#pragma unroll
      for (int ni = 0; ni < 4; ++ni)
        acc[mi][ni] = WMMA_F16(af[mi], bf[ni], acc[mi][ni]);
  }

#pragma unroll
  for (int mi = 0; mi < 2; ++mi)
#pragma unroll
    for (int ni = 0; ni < 4; ++ni)
#pragma unroll
      for (int j = 0; j < 8; ++j) {
        int row = m0 + wm * 32 + mi * 16 + mgrp + j;
        int col = n0 + wn * 64 + ni * 16 + (lane & 15);
        float v = acc[mi][ni][j] + bias[col];
        size_t idx = (size_t)row * N + col;
        if constexpr (OUTF) ((float*)outPtr)[idx] = v;
        else                ((_Float16*)outPtr)[idx] = (_Float16)v;
      }
}

// ---------------------------------------------------------------------------
// Flash attention. qkv layout: [b*T + t][3072] f16, cols = {q,k,v} x head x d.
// Grid: (T/128, B*H). 256 threads = 8 waves; wave w owns 16 query rows.
// 64-key blocks, double-buffered K/V staging (K async), online softmax.
// Row sums computed with a ones-matrix WMMA (P @ 1) instead of a lane
// butterfly; row max reduced with permlane16 on the VALU pipe.
// ---------------------------------------------------------------------------
#define ATT_T   2048
#define ATT_C3  3072
#define ATT_C   1024

__global__ __launch_bounds__(256)
void attn_kernel(const _Float16* __restrict__ qkv, _Float16* __restrict__ out) {
  __shared__ _Float16 Ks[2][64][64];      // [key][d]   16 KB
  __shared__ _Float16 Vt[2][64][64];      // [d][key]   16 KB
  __shared__ _Float16 Ps[8][16][64];      // per-wave P tile, 16 KB

  const int tid  = threadIdx.x;
  const int lane = tid & 31;
  const int w    = tid >> 5;
  const int bh   = blockIdx.y;
  const int b    = bh >> 4;
  const int h    = bh & 15;
  const int q0   = blockIdx.x * 128 + w * 16;
  const size_t rowBase = (size_t)b * ATT_T;

  const int mloc = lane & 15;
  const int koff = (lane >> 4) << 3;
  const int kb16 = (lane >> 4) << 4;
  const int mgrp = (lane >> 4) << 3;
  const float scale = 0.125f;             // 1/sqrt(64)

  // Resident Q fragments (16 rows x 64 d), pre-scaled.
  v16h qf[2];
  {
    const _Float16* qrow = qkv + (rowBase + q0 + mloc) * ATT_C3 + h * 64;
#pragma unroll
    for (int kc = 0; kc < 2; ++kc) {
      v16h t = load_frag16(qrow + kc * 32 + kb16, qrow + kc * 32 + kb16 + 8);
#pragma unroll
      for (int e = 0; e < 16; ++e) t[e] = (_Float16)((float)t[e] * scale);
      qf[kc] = t;
    }
  }

  // Ones B-matrix for WMMA row sums.
  v16h ones;
#pragma unroll
  for (int e = 0; e < 16; ++e) ones[e] = (_Float16)1.0f;

  float mrun[8], lrun[8];
  v8f zero = {};
  v8f oacc[4];
#pragma unroll
  for (int j = 0; j < 8; ++j) { mrun[j] = -1e30f; lrun[j] = 0.0f; }
#pragma unroll
  for (int dt = 0; dt < 4; ++dt) oacc[dt] = zero;

  auto stage = [&](int kblk, int buf) {
#pragma unroll
    for (int i = 0; i < 2; ++i) {        // K block: 64x64 = 512 chunks, async
      int c = tid + i * 256;
      int r = c >> 3, dc = (c & 7) << 3;
      async_copy16(&Ks[buf][r][dc],
                   &qkv[(rowBase + kblk + r) * ATT_C3 + 1024 + h * 64 + dc]);
    }
#pragma unroll
    for (int i = 0; i < 2; ++i) {        // V block transposed into [d][key]
      int c = tid + i * 256;
      int r = c >> 3, dc = (c & 7) << 3;
      v8h t = *(const v8h*)&qkv[(rowBase + kblk + r) * ATT_C3 + 2048 + h * 64 + dc];
#pragma unroll
      for (int e = 0; e < 8; ++e) Vt[buf][dc + e][r] = t[e];
    }
  };

  stage(0, 0);
  int buf = 0;
  for (int kblk = 0; kblk < ATT_T; kblk += 64, buf ^= 1) {
    async_join();
    __syncthreads();
    if (kblk + 64 < ATT_T) stage(kblk + 64, buf ^ 1);

    // Scores: S(16q x 64k), four 16x16 tiles, contraction over d.
    v8f S[4];
#pragma unroll
    for (int nt = 0; nt < 4; ++nt) {
      v8f s = zero;
#pragma unroll
      for (int kc = 0; kc < 2; ++kc) {
        const _Float16* r = &Ks[buf][nt * 16 + mloc][kc * 32 + kb16];
        v16h bfr = load_frag16(r, r + 8);
        s = WMMA_F16(qf[kc], bfr, s);
      }
      S[nt] = s;
    }

    // Row max across 4 tiles, then across the 16-lane half-wave (VALU permlane).
    float rmax[8];
#pragma unroll
    for (int j = 0; j < 8; ++j) {
      float r = fmaxf(fmaxf(S[0][j], S[1][j]), fmaxf(S[2][j], S[3][j]));
      r = fmaxf(r, xor16<1>(r));
      r = fmaxf(r, xor16<2>(r));
      r = fmaxf(r, xor16<4>(r));
      r = fmaxf(r, xor16<8>(r));
      rmax[j] = r;
    }

    float alpha[8];
#pragma unroll
    for (int j = 0; j < 8; ++j) {
      float mn = fmaxf(mrun[j], rmax[j]);
      alpha[j] = __expf(mrun[j] - mn);
      mrun[j]  = mn;
    }

    // P = exp(S - m): stash f16 P per-wave (C-layout -> A-layout relayout).
#pragma unroll
    for (int nt = 0; nt < 4; ++nt)
#pragma unroll
      for (int j = 0; j < 8; ++j) {
        float p = __expf(S[nt][j] - mrun[j]);
        Ps[w][mgrp + j][nt * 16 + (lane & 15)] = (_Float16)p;
      }

    // Reload P as A-fragments (also used for the WMMA row-sum).
    v16h pa[2];
#pragma unroll
    for (int kc = 0; kc < 2; ++kc) {
      const _Float16* pr = &Ps[w][mloc][kc * 32];
      pa[kc] = load_frag16(pr + kb16, pr + kb16 + 8);
    }

    // Row sums: P @ ones -> each row's sum replicated across the 16 lanes,
    // landing in exactly the lrun[] layout (C-frag row j <-> VGPR j).
    v8f ps = zero;
    ps = WMMA_F16(pa[0], ones, ps);
    ps = WMMA_F16(pa[1], ones, ps);
#pragma unroll
    for (int j = 0; j < 8; ++j) lrun[j] = lrun[j] * alpha[j] + ps[j];

#pragma unroll
    for (int dt = 0; dt < 4; ++dt)
#pragma unroll
      for (int j = 0; j < 8; ++j) oacc[dt][j] *= alpha[j];

    // O += P @ V (contraction over 64 keys).
#pragma unroll
    for (int kc = 0; kc < 2; ++kc) {
#pragma unroll
      for (int dt = 0; dt < 4; ++dt) {
        const _Float16* r = &Vt[buf][dt * 16 + mloc][kc * 32 + kb16];
        v16h vb = load_frag16(r, r + 8);
        oacc[dt] = WMMA_F16(pa[kc], vb, oacc[dt]);
      }
    }
  }

  // Normalize and emit f16 rows for the projection GEMM.
#pragma unroll
  for (int dt = 0; dt < 4; ++dt)
#pragma unroll
    for (int j = 0; j < 8; ++j) {
      int row = q0 + mgrp + j;
      float v = oacc[dt][j] / lrun[j];
      out[(rowBase + row) * ATT_C + h * 64 + dt * 16 + (lane & 15)] = (_Float16)v;
    }
}

// ---------------------------------------------------------------------------
// Host-side orchestration
// ---------------------------------------------------------------------------
extern "C" void kernel_launch(void* const* d_in, const int* in_sizes, int n_in,
                              void* d_out, int out_size, void* d_ws, size_t ws_size,
                              hipStream_t stream) {
  const float* x      = (const float*)d_in[0];  // [2,2048,1024]
  const float* w_qkv  = (const float*)d_in[1];  // [1024,3072]
  const float* b_qkv  = (const float*)d_in[2];  // [3072]
  const float* w_proj = (const float*)d_in[3];  // [1024,1024]
  const float* b_proj = (const float*)d_in[4];  // [1024]
  float* out = (float*)d_out;                   // [2,2048,1024]

  const int M = 4096;   // B*T
  const int C = 1024;
  const int C3 = 3072;

  _Float16* xh     = (_Float16*)d_ws;
  _Float16* wqkvT  = xh + (size_t)M * C;          // [3072][1024]
  _Float16* wprojT = wqkvT + (size_t)C3 * C;      // [1024][1024]
  _Float16* qkvh   = wprojT + (size_t)C * C;      // [4096][3072]
  _Float16* attnh  = qkvh + (size_t)M * C3;       // [4096][1024]

  // Staging: f16 activations, transposed f16 weights ([N][K]).
  cvt_f32_to_f16<<<4096, 256, 0, stream>>>(x, xh, M * C);
  cvt_transpose_f16<<<dim3(C3 / 32, C / 32), 256, 0, stream>>>(w_qkv, wqkvT, C, C3);
  cvt_transpose_f16<<<dim3(C / 32, C / 32), 256, 0, stream>>>(w_proj, wprojT, C, C);

  // QKV projection: [4096,1024] @ [1024,3072] -> f16 qkv
  gemm_wmma<1024, 3072, false><<<dim3(C3 / 128, M / 128), 256, 0, stream>>>(
      xh, wqkvT, b_qkv, qkvh);

  // Flash attention: 16 heads x 2 batches x 128-query tiles
  attn_kernel<<<dim3(ATT_T / 128, 2 * 16), 256, 0, stream>>>(qkvh, attnh);

  // Output projection: [4096,1024] @ [1024,1024] -> f32 out (+bias)
  gemm_wmma<1024, 1024, true><<<dim3(C / 128, M / 128), 256, 0, stream>>>(
      attnh, wprojT, b_proj, out);
}
